// GCN_63780264345860
// MI455X (gfx1250) — matrix operand order repaired and checked by reading, not compiled, not run on hardware
//
#include <hip/hip_runtime.h>
#include <hip/hip_bf16.h>

typedef __attribute__((ext_vector_type(16))) __bf16 bf16x16;
typedef __attribute__((ext_vector_type(8)))  float  f32x8;

#define NN  50000
#define EE  800000
#define FF  256
#define HH  256
#define LL  4
#define GG  64
#define NOTES_D 128

// ---------------------------------------------------------------------------
// generic zero
__global__ void zero_f(float* __restrict__ p, long n) {
    long i = (long)blockIdx.x * blockDim.x + threadIdx.x;
    long stride = (long)gridDim.x * blockDim.x;
    for (; i < n; i += stride) p[i] = 0.0f;
}

// ---------------------------------------------------------------------------
// pack a K x Ncols fp32 weight matrix into bf16 WMMA-B fragment-major layout:
// fragment index = ((kt * (Ncols/16) + nt) * 32 + lane), 16 bf16 per fragment.
// element j of fragment: B[kt*32 + ((j>=8)?16:0) + ((lane>=16)?8:0) + (j&7)][nt*16 + (lane&15)]
__global__ void pack_b(const float* __restrict__ W, __bf16* __restrict__ out,
                       int K, int Ncols) {
    long t = (long)blockIdx.x * blockDim.x + threadIdx.x;
    if (t >= (long)K * Ncols) return;
    long frag = t >> 4;
    int  j    = (int)(t & 15);
    int  lane = (int)(frag & 31);
    long tile = frag >> 5;
    int  ntiles = Ncols >> 4;
    int  nt = (int)(tile % ntiles);
    int  kt = (int)(tile / ntiles);
    int  k  = (kt << 5) + ((j >> 3) << 4) + ((lane >> 4) << 3) + (j & 7);
    int  n  = (nt << 4) + (lane & 15);
    out[t] = (__bf16)W[(long)k * Ncols + n];
}

// ---------------------------------------------------------------------------
// per-column sum / sumsq over rows (Ncols == 256 == blockDim.x)
__global__ void colstats(const float* __restrict__ X, int Nrows,
                         float* __restrict__ sum, float* __restrict__ sumsq) {
    int col = threadIdx.x;
    float s = 0.0f, s2 = 0.0f;
    for (long r = blockIdx.x; r < Nrows; r += gridDim.x) {
        float v = X[r * 256 + col];
        s += v; s2 += v * v;
    }
    atomicAdd(&sum[col], s);
    atomicAdd(&sumsq[col], s2);
}

__global__ void finalize_norm(const float* __restrict__ sum,
                              const float* __restrict__ sumsq,
                              const float* __restrict__ g,
                              const float* __restrict__ b,
                              float* __restrict__ scale,
                              float* __restrict__ shift, float invN) {
    int c = threadIdx.x;
    float mu  = sum[c] * invN;
    float var = sumsq[c] * invN - mu * mu;
    float s   = g[c] * rsqrtf(var + 1e-5f);
    scale[c] = s;
    shift[c] = b[c] - mu * s;
}

// ---------------------------------------------------------------------------
// WMMA GEMM: C[M x Ncols] = epilog(A' @ W + bias)
//  AMODE 0: A' = A0 * ascale[col] + ashift[col]   (fused input BatchNorm)
//  AMODE 1: A' = A0 + A1                          (fused GIN h + agg)
//  AMODE 2: A' = A0
//  DORELU 1: relu epilogue
// Each wave computes a 16(M) x 64(N) tile; K stepped by 32 with
// v_wmma_f32_16x16x32_bf16; A converted fp32->bf16 on the fly; B pre-packed.
template <int AMODE, int DORELU>
__launch_bounds__(256)
__global__ void gemm_bf16_wmma(const float* __restrict__ A0,
                               const float* __restrict__ A1,
                               const float* __restrict__ ascale,
                               const float* __restrict__ ashift,
                               const __bf16* __restrict__ Wp,
                               const float* __restrict__ bias,
                               float* __restrict__ C,
                               int M, int K, int Ncols) {
    const int lane = threadIdx.x & 31;
    const int wave = threadIdx.x >> 5;
    const int nChunks = Ncols >> 6;                  // 64-wide N chunks
    const long gw = (long)blockIdx.x * 8 + wave;
    const long rb = gw / nChunks;
    const int  nc = (int)(gw - rb * nChunks);
    if (rb >= (M >> 4)) return;                      // wave-uniform exit

    const int hi  = lane >> 4;                       // lane half (0/1)
    const int row = (int)rb * 16 + (lane & 15);      // A row this lane loads
    const int colBase = nc * 64;
    const int col = lane & 15;
    const int ntilesTotal = Ncols >> 4;

    const float2* Ar0 = reinterpret_cast<const float2*>(A0 + (long)row * K);
    const float2* Ar1 = reinterpret_cast<const float2*>(A1 ? A1 + (long)row * K : A0);
    const float2* Sc  = reinterpret_cast<const float2*>(ascale ? ascale : A0);
    const float2* Sh  = reinterpret_cast<const float2*>(ashift ? ashift : A0);
    const bf16x16* Bp = reinterpret_cast<const bf16x16*>(Wp);

    f32x8 acc[4] = {};

    const int kTiles = K >> 5;
    for (int kt = 0; kt < kTiles; ++kt) {
        // --- A fragment (16x32 bf16), per ISA 16-bit A layout ---
        bf16x16 a;
#pragma unroll
        for (int i = 0; i < 8; ++i) {
            int k = (kt << 5) + ((i & 4) ? 16 : 0) + (hi << 3) + ((i & 3) << 1);
            float2 v = Ar0[k >> 1];
            if (AMODE == 1) {
                float2 w = Ar1[k >> 1];
                v.x += w.x; v.y += w.y;
            } else if (AMODE == 0) {
                float2 s = Sc[k >> 1], t = Sh[k >> 1];
                v.x = v.x * s.x + t.x;
                v.y = v.y * s.y + t.y;
            }
            a[2 * i]     = (__bf16)v.x;
            a[2 * i + 1] = (__bf16)v.y;
        }
        // --- 4 pre-packed B fragments (32x16 each) ---
        long base = ((long)kt * ntilesTotal + (colBase >> 4)) * 32 + lane;
        bf16x16 bfr[4];
#pragma unroll
        for (int t = 0; t < 4; ++t) bfr[t] = Bp[base + t * 32];
#pragma unroll
        for (int t = 0; t < 4; ++t)
            acc[t] = __builtin_amdgcn_wmma_f32_16x16x32_bf16(
                false, a, false, bfr[t], (short)0, acc[t], false, false);
    }

    // --- epilogue: bias (+relu), store fp32 ---
    const int outRow0 = (int)rb * 16 + (hi << 3);
#pragma unroll
    for (int t = 0; t < 4; ++t) {
        int cidx = colBase + t * 16 + col;
        float bv = bias[cidx];
#pragma unroll
        for (int j = 0; j < 8; ++j) {
            float v = acc[t][j] + bv;
            if (DORELU) v = fmaxf(v, 0.0f);
            C[(long)(outRow0 + j) * Ncols + cidx] = v;
        }
    }
}

// ---------------------------------------------------------------------------
// scatter-add: agg[dst[e], :] += h[src[e], :]   (8 edges per 256-thread block)
__global__ void scatter_edges(const float* __restrict__ h,
                              const int* __restrict__ src,
                              const int* __restrict__ dst,
                              float* __restrict__ agg, int nE) {
    int f = threadIdx.x;
    long e0 = (long)blockIdx.x * 8;
#pragma unroll
    for (int i = 0; i < 8; ++i) {
        long e = e0 + i;
        if (e >= nE) break;
        int s = src[e], d = dst[e];
        atomicAdd(&agg[(long)d * 256 + f], h[(long)s * 256 + f]);
    }
}

// ---------------------------------------------------------------------------
// h += relu(m * scale[col] + shift[col])   (fused BN + relu + residual)
__global__ void bn_relu_residual(const float* __restrict__ m,
                                 const float* __restrict__ scale,
                                 const float* __restrict__ shift,
                                 float* __restrict__ h, long n) {
    long i = (long)blockIdx.x * blockDim.x + threadIdx.x;
    long stride = (long)gridDim.x * blockDim.x;
    for (; i < n; i += stride) {
        int c = (int)(i & 255);
        float v = m[i] * scale[c] + shift[c];
        h[i] += fmaxf(v, 0.0f);
    }
}

// ---------------------------------------------------------------------------
// per-graph mean pool
__global__ void pool_sum(const float* __restrict__ h, const int* __restrict__ batch,
                         float* __restrict__ psum, float* __restrict__ pcnt) {
    int f = threadIdx.x;
    long nd = blockIdx.x;
    int g = batch[nd];
    atomicAdd(&psum[(long)g * 256 + f], h[nd * 256 + f]);
    if (f == 0) atomicAdd(&pcnt[g], 1.0f);
}

__global__ void pool_finalize(const float* __restrict__ psum,
                              const float* __restrict__ pcnt,
                              float* __restrict__ embed) {
    int g = blockIdx.x, f = threadIdx.x;
    float c = fmaxf(pcnt[g], 1.0f);
    embed[(long)g * 256 + f] = psum[(long)g * 256 + f] / c;
}

// logits = embed @ pred_W + pred_b   (64x256x128, tiny)
__global__ void logits_k(const float* __restrict__ embed,
                         const float* __restrict__ W,
                         const float* __restrict__ b,
                         float* __restrict__ out) {
    int g = blockIdx.x, n = threadIdx.x;   // 128 threads
    float acc = b[n];
    for (int k = 0; k < 256; ++k)
        acc += embed[(long)g * 256 + k] * W[(long)k * 128 + n];
    out[(long)g * 128 + n] = acc;
}

// ---------------------------------------------------------------------------
extern "C" void kernel_launch(void* const* d_in, const int* in_sizes, int n_in,
                              void* d_out, int out_size, void* d_ws, size_t ws_size,
                              hipStream_t stream) {
    const float* x      = (const float*)d_in[0];
    const int*   ei     = (const int*)d_in[1];
    const int*   batch  = (const int*)d_in[2];
    const float* fn_g   = (const float*)d_in[3];
    const float* fn_b   = (const float*)d_in[4];
    const float* proj_W = (const float*)d_in[5];
    const float* proj_b = (const float*)d_in[6];
    const float* W1     = (const float*)d_in[7];
    const float* b1     = (const float*)d_in[8];
    const float* W2     = (const float*)d_in[9];
    const float* b2     = (const float*)d_in[10];
    const float* bn_g   = (const float*)d_in[11];
    const float* bn_b   = (const float*)d_in[12];
    const float* pred_W = (const float*)d_in[13];
    const float* pred_b = (const float*)d_in[14];
    float* out = (float*)d_out;

    float* ws = (float*)d_ws;
    const long NH = (long)NN * HH;            // 12.8M floats
    float* h    = ws;                          // current node features
    float* agg  = ws + NH;                     // neighbor sums; reused as m
    float* t1   = ws + 2 * NH;                 // MLP hidden
    float* stat = ws + 3 * NH;
    float* colsum   = stat;        float* colsumsq = stat + 256;
    float* fscale   = stat + 512;  float* fshift   = stat + 768;
    float* bnsum    = stat + 1024; float* bnsumsq  = stat + 1280;
    float* bnscale  = stat + 1536; float* bnshift  = stat + 1792;
    float* psum     = stat + 2048;             // G*H = 16384
    float* pcnt     = psum + (long)GG * HH;    // 64
    __bf16* projWp = (__bf16*)(pcnt + GG);     // 32B-aligned by construction
    __bf16* W1p    = projWp + 65536;
    __bf16* W2p    = W1p + 4 * 65536;

    // zero all small accumulators (colsum..bnshift, psum, pcnt)
    zero_f<<<128, 256, 0, stream>>>(stat, 2048 + (long)GG * HH + GG);

    // pack weights to bf16 fragment-major
    pack_b<<<256, 256, 0, stream>>>(proj_W, projWp, FF, HH);
    for (int l = 0; l < LL; ++l) {
        pack_b<<<256, 256, 0, stream>>>(W1 + (long)l * 65536, W1p + (long)l * 65536, HH, HH);
        pack_b<<<256, 256, 0, stream>>>(W2 + (long)l * 65536, W2p + (long)l * 65536, HH, HH);
    }

    // input feature BatchNorm stats + fused projection GEMM
    colstats<<<512, 256, 0, stream>>>(x, NN, colsum, colsumsq);
    finalize_norm<<<1, 256, 0, stream>>>(colsum, colsumsq, fn_g, fn_b,
                                         fscale, fshift, 1.0f / NN);

    const int gemmBlocks = ((NN / 16) * (HH / 64) + 7) / 8;   // 1563
    gemm_bf16_wmma<0, 1><<<gemmBlocks, 256, 0, stream>>>(
        x, nullptr, fscale, fshift, projWp, proj_b, h, NN, FF, HH);

    // GIN blocks
    for (int l = 0; l < LL; ++l) {
        zero_f<<<4096, 256, 0, stream>>>(agg, NH);
        zero_f<<<2, 256, 0, stream>>>(bnsum, 512);
        scatter_edges<<<EE / 8, 256, 0, stream>>>(h, ei, ei + EE, agg, EE);
        gemm_bf16_wmma<1, 1><<<gemmBlocks, 256, 0, stream>>>(
            h, agg, nullptr, nullptr, W1p + (long)l * 65536, b1 + l * HH,
            t1, NN, HH, HH);
        gemm_bf16_wmma<2, 0><<<gemmBlocks, 256, 0, stream>>>(
            t1, nullptr, nullptr, nullptr, W2p + (long)l * 65536, b2 + l * HH,
            agg /* = m */, NN, HH, HH);
        colstats<<<512, 256, 0, stream>>>(agg, NN, bnsum, bnsumsq);
        finalize_norm<<<1, 256, 0, stream>>>(bnsum, bnsumsq, bn_g + l * HH,
                                             bn_b + l * HH, bnscale, bnshift,
                                             1.0f / NN);
        bn_relu_residual<<<4096, 256, 0, stream>>>(agg, bnscale, bnshift, h, NH);
    }

    // readout
    pool_sum<<<NN, 256, 0, stream>>>(h, batch, psum, pcnt);
    pool_finalize<<<GG, 256, 0, stream>>>(psum, pcnt, out);
    logits_k<<<GG, NOTES_D, 0, stream>>>(out, pred_W, pred_b, out + (long)GG * HH);
}
